// Attention_29506425323809
// MI455X (gfx1250) — matrix operand order repaired and checked by reading, not compiled, not run on hardware
//
#include <hip/hip_runtime.h>
#include <hip/hip_bf16.h>

// ---- problem constants (from reference) ----
#define BB   2
#define NN   2048
#define DD   1024
#define HH   16
#define DH   64

typedef __attribute__((ext_vector_type(16))) _Float16 v16h;
typedef __attribute__((ext_vector_type(8)))  _Float16 v8h;
typedef __attribute__((ext_vector_type(4)))  _Float16 v4h;
typedef __attribute__((ext_vector_type(2)))  _Float16 v2h;
typedef __attribute__((ext_vector_type(8)))  float    f32x8;
typedef __attribute__((ext_vector_type(4)))  float    f32x4;

__device__ __forceinline__ f32x8 wmma_f16(v16h a, v16h b, f32x8 c) {
    // (neg_a, A, neg_b, B, c_mod, C, reuse_a, reuse_b)
    return __builtin_amdgcn_wmma_f32_16x16x32_f16(false, a, false, b,
                                                  (short)0, c, false, false);
}

// Async global -> LDS copy (CDNA5, tracked by ASYNCcnt).
// lds_off: byte offset of destination within the block's LDS allocation
// (single __shared__ object => allocation starts at LDS offset 0).
__device__ __forceinline__ void async_load_b128(unsigned lds_off, const void* g) {
    asm volatile("global_load_async_to_lds_b128 %0, %1, off"
                 :: "v"(lds_off), "v"((unsigned long long)(size_t)g)
                 : "memory");
}
__device__ __forceinline__ void wait_async0() {
    asm volatile("s_wait_asynccnt 0" ::: "memory");
}

// Load a 16-half fragment as two contiguous 16-byte LDS chunks.
//   A-fragment (tile [m][k]): o1 = kb + 8*half,  o2 = kb + 16 + 8*half
//   B-fragment (tile [j][k]): o1 = kb + 8*half,  o2 = kb + 16 + 8*half
// Both give the same (table) k order.
__device__ __forceinline__ v16h frag2(const _Float16* row, int o1, int o2) {
    v8h lo = *(const v8h*)(row + o1);
    v8h hi = *(const v8h*)(row + o2);
    return __builtin_shufflevector(lo, hi,
                                   0,1,2,3,4,5,6,7,8,9,10,11,12,13,14,15);
}

__device__ __forceinline__ v4h cvt4(f32x4 f) {
    v4h h;
    h[0] = (_Float16)f[0]; h[1] = (_Float16)f[1];
    h[2] = (_Float16)f[2]; h[3] = (_Float16)f[3];
    return h;
}

// ============================================================
// Kernel 1: QKV projection + bias + RoPE, write f16 [B,H,N,64]
// grid (32, 16) = (rows/128, D/64); block 256 (8 waves)
// ============================================================
__global__ __launch_bounds__(256)
void qkv_rope_kernel(const float* __restrict__ x,
                     const float* __restrict__ cosb,
                     const float* __restrict__ sinb,
                     const float* __restrict__ Wq, const float* __restrict__ bq,
                     const float* __restrict__ Wk, const float* __restrict__ bk,
                     const float* __restrict__ Wv, const float* __restrict__ bv,
                     _Float16* __restrict__ Qh,
                     _Float16* __restrict__ Kh,
                     _Float16* __restrict__ Vh)
{
    __shared__ _Float16 As[128][40];     // x tile [m][k]  (stride 80B: conflict-free b128)
    __shared__ _Float16 Bs[3][64][40];   // W tiles [mat][j][k] (W natural row-major)

    const int tid  = threadIdx.x;
    const int wave = tid >> 5;
    const int lane = tid & 31;
    const int half = lane >> 4;
    const int l16  = lane & 15;
    const int row0 = blockIdx.x * 128;   // global row in [0, B*N)
    const int col0 = blockIdx.y * 64;    // output feature base; head = blockIdx.y

    const f32x8 fz = {0.f,0.f,0.f,0.f,0.f,0.f,0.f,0.f};
    f32x8 accq[4] = {fz,fz,fz,fz};
    f32x8 acck[4] = {fz,fz,fz,fz};
    f32x8 accv[4] = {fz,fz,fz,fz};

    const int am = wave * 16 + l16;      // A-fragment row (block-local)

    for (int kb = 0; kb < DD; kb += 32) {
        __syncthreads();
        // stage x tile (128x32): float4 load -> v4h store
        for (int i = tid; i < 128 * 8; i += 256) {
            int m = i >> 3, k4 = (i & 7) * 4;
            f32x4 f = *(const f32x4*)&x[(size_t)(row0 + m) * DD + kb + k4];
            *(v4h*)&As[m][k4] = cvt4(f);
        }
        // stage 3 weight tiles as Bt[j][k] = W[col0+j][kb+k] (natural order)
        for (int i = tid; i < 64 * 8; i += 256) {
            int j = i >> 3, k4 = (i & 7) * 4;
            size_t wi = (size_t)(col0 + j) * DD + kb + k4;
            *(v4h*)&Bs[0][j][k4] = cvt4(*(const f32x4*)&Wq[wi]);
            *(v4h*)&Bs[1][j][k4] = cvt4(*(const f32x4*)&Wk[wi]);
            *(v4h*)&Bs[2][j][k4] = cvt4(*(const f32x4*)&Wv[wi]);
        }
        __syncthreads();

        v16h a = frag2(&As[am][0], 8 * half, 16 + 8 * half);
        // process two column-tiles at a time: 6 live B-fragments before the
        // WMMA group -> scheduler can overlap ds_load_b128 with wmma
        #pragma unroll
        for (int jp = 0; jp < 4; jp += 2) {
            const int j0 = jp * 16 + l16;
            const int j1 = (jp + 1) * 16 + l16;
            v16h b00 = frag2(&Bs[0][j0][0], 8 * half, 16 + 8 * half);
            v16h b01 = frag2(&Bs[1][j0][0], 8 * half, 16 + 8 * half);
            v16h b02 = frag2(&Bs[2][j0][0], 8 * half, 16 + 8 * half);
            v16h b10 = frag2(&Bs[0][j1][0], 8 * half, 16 + 8 * half);
            v16h b11 = frag2(&Bs[1][j1][0], 8 * half, 16 + 8 * half);
            v16h b12 = frag2(&Bs[2][j1][0], 8 * half, 16 + 8 * half);
            accq[jp]     = wmma_f16(a, b00, accq[jp]);
            acck[jp]     = wmma_f16(a, b01, acck[jp]);
            accv[jp]     = wmma_f16(a, b02, accv[jp]);
            accq[jp + 1] = wmma_f16(a, b10, accq[jp + 1]);
            acck[jp + 1] = wmma_f16(a, b11, acck[jp + 1]);
            accv[jp + 1] = wmma_f16(a, b12, accv[jp + 1]);
        }
    }

    // epilogue: bias + RoPE (adjacent columns live in lanes L, L^1),
    // pair-packed v2h stores from even lanes, layout [B,H,N,64] f16
    const int h = blockIdx.y;
    #pragma unroll
    for (int jt = 0; jt < 4; ++jt) {
        #pragma unroll
        for (int r = 0; r < 8; ++r) {
            int rg = row0 + wave * 16 + r + 8 * half;   // global row (b*N+n)
            int cg = col0 + jt * 16 + l16;              // global feature
            int b  = rg >> 11;                          // / N
            int n  = rg & (NN - 1);
            int dh = jt * 16 + l16;
            float cs = cosb[(size_t)rg * DD + cg];
            float sn = sinb[(size_t)rg * DD + cg];
            size_t oidx = (((size_t)b * HH + h) * NN + n) * DH + dh;

            float tq = accq[jt][r] + bq[cg];
            float pq = __shfl_xor(tq, 1, 32);
            float rotq = (lane & 1) ? pq : -pq;   // even: -t[odd], odd: +t[even]
            float qv = tq * cs + rotq * sn;

            float tk = acck[jt][r] + bk[cg];
            float pk = __shfl_xor(tk, 1, 32);
            float rotk = (lane & 1) ? pk : -pk;
            float kv = tk * cs + rotk * sn;

            float vv = accv[jt][r] + bv[cg];

            float qn = __shfl_xor(qv, 1, 32);
            float kn = __shfl_xor(kv, 1, 32);
            float vn = __shfl_xor(vv, 1, 32);
            if (!(lane & 1)) {
                v2h q2; q2[0] = (_Float16)qv; q2[1] = (_Float16)qn;
                v2h k2; k2[0] = (_Float16)kv; k2[1] = (_Float16)kn;
                v2h vh2; vh2[0] = (_Float16)vv; vh2[1] = (_Float16)vn;
                *(v2h*)&Qh[oidx] = q2;
                *(v2h*)&Kh[oidx] = k2;
                *(v2h*)&Vh[oidx] = vh2;
            }
        }
    }
}

// ============================================================
// Kernel 2: flash attention, one (b,h) x 64-row q tile per block
// grid (B*H, N/64) = (32, 32); block 128 (4 waves)
// LDS laid out in one block so async-to-LDS offsets are explicit.
// ============================================================
#define QS_OFF  0u
#define KS_OFF  9216u
#define VST_OFF 18432u
#define PS_OFF  27648u

__global__ __launch_bounds__(128)
void attn_kernel(const _Float16* __restrict__ Qh,
                 const _Float16* __restrict__ Kh,
                 const _Float16* __restrict__ Vh,
                 _Float16* __restrict__ ctxh)
{
    __shared__ __align__(16) char smem[4 * 64 * 72 * 2];
    _Float16 (*Qs)[72]  = (_Float16(*)[72])(smem + QS_OFF);   // [m][dh]
    _Float16 (*Ks)[72]  = (_Float16(*)[72])(smem + KS_OFF);   // [key][dh]
    _Float16 (*Vst)[72] = (_Float16(*)[72])(smem + VST_OFF);  // [dh][key]
    _Float16 (*Ps)[72]  = (_Float16(*)[72])(smem + PS_OFF);   // [m][key]

    const int bh   = blockIdx.x;             // b*H + h
    const int q0   = blockIdx.y * 64;
    const int tid  = threadIdx.x;
    const int wave = tid >> 5;
    const int lane = tid & 31;
    const int half = lane >> 4;
    const int l16  = lane & 15;
    const size_t base = (size_t)bh * NN * DH;

    // async-stage the Q tile (pure copy, f16)
    for (int i = tid; i < 64 * 8; i += 128) {
        int m = i >> 3, d8 = (i & 7) * 8;
        async_load_b128(QS_OFF + (unsigned)(m * 72 + d8) * 2,
                        &Qh[base + (size_t)(q0 + m) * DH + d8]);
    }
    wait_async0();
    __syncthreads();

    const int am = wave * 16 + l16;
    v16h aq[2];                               // Q A-fragments, fixed all loop
    #pragma unroll
    for (int ks = 0; ks < 2; ++ks)
        aq[ks] = frag2(&Qs[am][0], 32 * ks + 8 * half, 32 * ks + 16 + 8 * half);

    const f32x8 fz = {0.f,0.f,0.f,0.f,0.f,0.f,0.f,0.f};
    f32x8 accO[4] = {fz,fz,fz,fz};
    float mrow[8], lrow[8];
    #pragma unroll
    for (int r = 0; r < 8; ++r) { mrow[r] = -3.0e38f; lrow[r] = 0.f; }

    for (int k0 = 0; k0 < NN; k0 += 64) {
        __syncthreads();
        // K tile: async global->LDS copy; V tile: load + transpose store
        for (int i = tid; i < 64 * 8; i += 128) {
            int m = i >> 3, d8 = (i & 7) * 8;
            async_load_b128(KS_OFF + (unsigned)(m * 72 + d8) * 2,
                            &Kh[base + (size_t)(k0 + m) * DH + d8]);
            v8h v = *(const v8h*)&Vh[base + (size_t)(k0 + m) * DH + d8];
            #pragma unroll
            for (int e = 0; e < 8; ++e) Vst[d8 + e][m] = v[e];  // transpose
        }
        wait_async0();
        __syncthreads();

        // S = Q * K^T : B[k=dh][j=key] -> rows of Ks (contiguous in dh)
        f32x8 accS[4] = {fz,fz,fz,fz};
        #pragma unroll
        for (int ks = 0; ks < 2; ++ks) {
            v16h bf0 = frag2(&Ks[0 * 16 + l16][0], 32 * ks + 8 * half, 32 * ks + 16 + 8 * half);
            v16h bf1 = frag2(&Ks[1 * 16 + l16][0], 32 * ks + 8 * half, 32 * ks + 16 + 8 * half);
            v16h bf2 = frag2(&Ks[2 * 16 + l16][0], 32 * ks + 8 * half, 32 * ks + 16 + 8 * half);
            v16h bf3 = frag2(&Ks[3 * 16 + l16][0], 32 * ks + 8 * half, 32 * ks + 16 + 8 * half);
            accS[0] = wmma_f16(aq[ks], bf0, accS[0]);
            accS[1] = wmma_f16(aq[ks], bf1, accS[1]);
            accS[2] = wmma_f16(aq[ks], bf2, accS[2]);
            accS[3] = wmma_f16(aq[ks], bf3, accS[3]);
        }

        // online softmax (row r lives in lanes of one half -> masks 1..8)
        #pragma unroll
        for (int r = 0; r < 8; ++r) {
            float mx = -3.0e38f;
            #pragma unroll
            for (int jt = 0; jt < 4; ++jt) mx = fmaxf(mx, accS[jt][r]);
            #pragma unroll
            for (int off = 8; off; off >>= 1)
                mx = fmaxf(mx, __shfl_xor(mx, off, 32));
            float mnew  = fmaxf(mrow[r], mx);
            float scale = __expf(mrow[r] - mnew);
            mrow[r] = mnew;
            float rsum = 0.f;
            int mloc = wave * 16 + r + 8 * half;
            #pragma unroll
            for (int jt = 0; jt < 4; ++jt) {
                float p  = __expf(accS[jt][r] - mnew);
                float pn = __shfl_xor(p, 1, 32);
                if (!(lane & 1)) {                 // pair-packed v2h store
                    v2h p2; p2[0] = (_Float16)p; p2[1] = (_Float16)pn;
                    *(v2h*)&Ps[mloc][jt * 16 + l16] = p2;
                }
                rsum += p;
            }
            #pragma unroll
            for (int off = 8; off; off >>= 1)
                rsum += __shfl_xor(rsum, off, 32);
            lrow[r] = lrow[r] * scale + rsum;
            #pragma unroll
            for (int jt = 0; jt < 4; ++jt) accO[jt][r] *= scale;
        }
        __syncthreads();   // order Ps/Vst traffic

        // O += P * V : A = own rows of Ps; B[k=key][j=dh] -> rows of Vst
        #pragma unroll
        for (int ks = 0; ks < 2; ++ks) {
            v16h ap = frag2(&Ps[am][0],
                            32 * ks + 8 * half, 32 * ks + 16 + 8 * half);
            v16h bf0 = frag2(&Vst[0 * 16 + l16][0], 32 * ks + 8 * half, 32 * ks + 16 + 8 * half);
            v16h bf1 = frag2(&Vst[1 * 16 + l16][0], 32 * ks + 8 * half, 32 * ks + 16 + 8 * half);
            v16h bf2 = frag2(&Vst[2 * 16 + l16][0], 32 * ks + 8 * half, 32 * ks + 16 + 8 * half);
            v16h bf3 = frag2(&Vst[3 * 16 + l16][0], 32 * ks + 8 * half, 32 * ks + 16 + 8 * half);
            accO[0] = wmma_f16(ap, bf0, accO[0]);
            accO[1] = wmma_f16(ap, bf1, accO[1]);
            accO[2] = wmma_f16(ap, bf2, accO[2]);
            accO[3] = wmma_f16(ap, bf3, accO[3]);
        }
    }

    // finalize: O / l, write ctx as f16 [B, N, H*64] (pair-packed)
    const int b = bh >> 4;
    const int h = bh & (HH - 1);
    #pragma unroll
    for (int jt = 0; jt < 4; ++jt)
        #pragma unroll
        for (int r = 0; r < 8; ++r) {
            int n  = q0 + wave * 16 + r + 8 * half;
            int dh = jt * 16 + l16;
            float o  = accO[jt][r] / lrow[r];
            float on = __shfl_xor(o, 1, 32);
            if (!(lane & 1)) {
                v2h o2; o2[0] = (_Float16)o; o2[1] = (_Float16)on;
                *(v2h*)&ctxh[((size_t)(b * NN + n)) * DD + h * DH + dh] = o2;
            }
        }
}

// ============================================================
// Kernel 3: output projection ctx @ Wo^T + bo -> f32
// grid (32, 16); block 256 (8 waves)
// ============================================================
#define OAS_OFF 0u
#define OBS_OFF 10240u

__global__ __launch_bounds__(256)
void oproj_kernel(const _Float16* __restrict__ ctxh,
                  const float* __restrict__ Wo,
                  const float* __restrict__ bo,
                  float* __restrict__ out)
{
    __shared__ __align__(16) char smem[128 * 40 * 2 + 64 * 40 * 2];
    _Float16 (*As)[40] = (_Float16(*)[40])(smem + OAS_OFF);  // ctx tile [m][k]
    _Float16 (*Bs)[40] = (_Float16(*)[40])(smem + OBS_OFF);  // Wo tile  [j][k]

    const int tid  = threadIdx.x;
    const int wave = tid >> 5;
    const int lane = tid & 31;
    const int half = lane >> 4;
    const int l16  = lane & 15;
    const int row0 = blockIdx.x * 128;
    const int col0 = blockIdx.y * 64;

    const f32x8 fz = {0.f,0.f,0.f,0.f,0.f,0.f,0.f,0.f};
    f32x8 acc[4] = {fz,fz,fz,fz};
    const int am = wave * 16 + l16;

    for (int kb = 0; kb < DD; kb += 32) {
        __syncthreads();
        // ctx A-tile: pure f16 copy -> async global->LDS
        for (int i = tid; i < 128 * 4; i += 256) {
            int m = i >> 2, k8 = (i & 3) * 8;
            async_load_b128(OAS_OFF + (unsigned)(m * 40 + k8) * 2,
                            &ctxh[(size_t)(row0 + m) * DD + kb + k8]);
        }
        for (int i = tid; i < 64 * 8; i += 256) {      // float4 -> v4h
            int j = i >> 3, k4 = (i & 7) * 4;
            f32x4 f = *(const f32x4*)&Wo[(size_t)(col0 + j) * DD + kb + k4];
            *(v4h*)&Bs[j][k4] = cvt4(f);
        }
        wait_async0();
        __syncthreads();

        v16h a = frag2(&As[am][0], 8 * half, 16 + 8 * half);
        v16h bf0 = frag2(&Bs[0 * 16 + l16][0], 8 * half, 16 + 8 * half);
        v16h bf1 = frag2(&Bs[1 * 16 + l16][0], 8 * half, 16 + 8 * half);
        v16h bf2 = frag2(&Bs[2 * 16 + l16][0], 8 * half, 16 + 8 * half);
        v16h bf3 = frag2(&Bs[3 * 16 + l16][0], 8 * half, 16 + 8 * half);
        acc[0] = wmma_f16(a, bf0, acc[0]);
        acc[1] = wmma_f16(a, bf1, acc[1]);
        acc[2] = wmma_f16(a, bf2, acc[2]);
        acc[3] = wmma_f16(a, bf3, acc[3]);
    }

    #pragma unroll
    for (int jt = 0; jt < 4; ++jt)
        #pragma unroll
        for (int r = 0; r < 8; ++r) {
            int rg = row0 + wave * 16 + r + 8 * half;
            int cg = col0 + jt * 16 + l16;
            out[(size_t)rg * DD + cg] = acc[jt][r] + bo[cg];
        }
}

// ============================================================
extern "C" void kernel_launch(void* const* d_in, const int* in_sizes, int n_in,
                              void* d_out, int out_size, void* d_ws, size_t ws_size,
                              hipStream_t stream)
{
    (void)in_sizes; (void)n_in; (void)out_size; (void)ws_size;
    const float* x    = (const float*)d_in[0];
    const float* cosb = (const float*)d_in[1];
    const float* sinb = (const float*)d_in[2];
    const float* Wq   = (const float*)d_in[3];
    const float* bq   = (const float*)d_in[4];
    const float* Wk   = (const float*)d_in[5];
    const float* bk   = (const float*)d_in[6];
    const float* Wv   = (const float*)d_in[7];
    const float* bv   = (const float*)d_in[8];
    const float* Wo   = (const float*)d_in[9];
    const float* bo   = (const float*)d_in[10];

    const size_t elems = (size_t)BB * NN * DD;   // 4M elems per buffer
    _Float16* Qh   = (_Float16*)d_ws;
    _Float16* Kh   = Qh + elems;
    _Float16* Vh   = Kh + elems;
    _Float16* ctxh = Vh + elems;

    dim3 g1((BB * NN) / 128, DD / 64);           // 32 x 16
    qkv_rope_kernel<<<g1, dim3(256), 0, stream>>>(x, cosb, sinb,
                                                  Wq, bq, Wk, bk, Wv, bv,
                                                  Qh, Kh, Vh);

    dim3 g2(BB * HH, NN / 64);                   // 32 x 32
    attn_kernel<<<g2, dim3(128), 0, stream>>>(Qh, Kh, Vh, ctxh);

    dim3 g3((BB * NN) / 128, DD / 64);           // 32 x 16
    oproj_kernel<<<g3, dim3(256), 0, stream>>>(ctxh, Wo, bo, (float*)d_out);
}
